// HeteroGNN_76690936037661
// MI455X (gfx1250) — compile-verified, matrix operand order checked by reading, not run on hardware
//
#include <hip/hip_runtime.h>
#include <hip/hip_bf16.h>
#include <math.h>

// ---------------------------------------------------------------------------
// HeteroGNN (2-layer GAT) for MI455X / gfx1250.
//  - Dense projections: bf16 WMMA (v_wmma_f32_16x16x32_bf16), fp32 accum,
//    double-buffered async global->LDS tile copies (ASYNCcnt) when available.
//  - Edge softmax aggregation: 3-pass segment softmax with hardware fp32
//    atomics (dst accumulators fit in the 192MB L2).
// ---------------------------------------------------------------------------

typedef __attribute__((ext_vector_type(16))) __bf16 v16bf;
typedef __attribute__((ext_vector_type(8)))  float  v8f;
typedef int v4i_ __attribute__((vector_size(16)));   // matches builtin param type

#define HIDC 256
#define BM 128
#define BN 128
#define BK 32
#define LDT 40          // LDS row stride in halves (80B: 16B-aligned, conflict-free-ish)
#define NEG_SLOPE 0.2f

#if __has_builtin(__builtin_amdgcn_global_load_async_to_lds_b128) && \
    __has_builtin(__builtin_amdgcn_s_wait_asynccnt)
#define USE_ASYNC_LDS 1
#else
#define USE_ASYNC_LDS 0
#endif

#if USE_ASYNC_LDS
// param1: global (AS1) int4*, param2: LDS (AS3) int4*, then imm offset, imm cpol
#define ASYNC_B128(gp, lp)                                                  \
    __builtin_amdgcn_global_load_async_to_lds_b128(                         \
        (__attribute__((address_space(1))) v4i_*)(gp),                      \
        (__attribute__((address_space(3))) v4i_*)(lp), 0, 0)
#endif

union Frag16 { unsigned int u[8]; uint4 q[2]; v16bf v; };
static_assert(sizeof(Frag16) == 32, "frag size");

// ----------------------------- GEMM (WMMA) ---------------------------------
// C[M,N] = A[M,K] (bf16, row-major) * B[N,K]^T (bf16, row-major) [+bias][relu]
// zpad: 256B zero-filled global scratch used as the read target for OOB rows.
__global__ __launch_bounds__(256)
void gemm_bf16_wmma(const unsigned short* __restrict__ A,
                    const unsigned short* __restrict__ B,
                    float* __restrict__ C,
                    const float* __restrict__ bias,
                    const unsigned short* __restrict__ zpad,
                    int M, int N, int K, int doRelu)
{
    const int tid   = threadIdx.x;
    const int lane  = tid & 31;
    const int wave  = tid >> 5;
    const int waveM = wave & 3;   // 4 waves along M
    const int waveN = wave >> 2;  // 2 waves along N
    const int bM = blockIdx.y * BM;
    const int bN = blockIdx.x * BN;

    v8f acc[2][4] = {};

    const int r   = tid >> 1;         // 0..127  (tile row handled by this thread)
    const int seg = (tid & 1) * 16;   // half-offset within row

    const int g  = lane >> 4;         // lane group (K +8 halves)
    const int lr = lane & 15;

#if USE_ASYNC_LDS
    // ---------------- async double-buffered pipeline ----------------
    __shared__ __align__(16) unsigned short As[2][BM * LDT];
    __shared__ __align__(16) unsigned short Bs[2][BN * LDT];

    const bool padA = (bM + r) >= M;
    const bool padB = (bN + r) >= N;
    const unsigned short* gA = padA ? zpad : (A + (size_t)(bM + r) * K + seg);
    const unsigned short* gB = padB ? zpad : (B + (size_t)(bN + r) * K + seg);

    auto issue = [&](int buf, int k0) {
        const unsigned short* ga = padA ? zpad : (gA + k0);
        ASYNC_B128(ga,     &As[buf][r * LDT + seg]);
        ASYNC_B128(ga + 8, &As[buf][r * LDT + seg + 8]);
        const unsigned short* gb = padB ? zpad : (gB + k0);
        ASYNC_B128(gb,     &Bs[buf][r * LDT + seg]);
        ASYNC_B128(gb + 8, &Bs[buf][r * LDT + seg + 8]);
    };

    const int nK = K / BK;
    issue(0, 0);
    for (int kt = 0; kt < nK; ++kt) {
        const int cur = kt & 1;
        if (kt + 1 < nK) {
            issue(cur ^ 1, (kt + 1) * BK);       // prefetch next tile (other buffer)
            __builtin_amdgcn_s_wait_asynccnt(4); // current tile's 4 copies complete
        } else {
            __builtin_amdgcn_s_wait_asynccnt(0);
        }
        __syncthreads();                          // all waves' copies visible

        Frag16 aF[2], bF[4];
#pragma unroll
        for (int i = 0; i < 2; ++i) {
            int row = waveM * 32 + i * 16 + lr;
            aF[i].q[0] = *(const uint4*)(&As[cur][row * LDT + 8 * g]);
            aF[i].q[1] = *(const uint4*)(&As[cur][row * LDT + 16 + 8 * g]);
        }
#pragma unroll
        for (int j = 0; j < 4; ++j) {
            int col = waveN * 64 + j * 16 + lr;
            bF[j].q[0] = *(const uint4*)(&Bs[cur][col * LDT + 8 * g]);
            bF[j].q[1] = *(const uint4*)(&Bs[cur][col * LDT + 16 + 8 * g]);
        }
#pragma unroll
        for (int i = 0; i < 2; ++i)
#pragma unroll
            for (int j = 0; j < 4; ++j)
                acc[i][j] = __builtin_amdgcn_wmma_f32_16x16x32_bf16(
                    false, aF[i].v, false, bF[j].v, (short)0, acc[i][j], false, false);

        __syncthreads();                          // reads done before buffer reuse
    }
#else
    // ---------------- fallback: sync loads staged through VGPRs ----------------
    (void)zpad;
    __shared__ __align__(16) unsigned short As[BM * LDT];
    __shared__ __align__(16) unsigned short Bs[BN * LDT];

    for (int k0 = 0; k0 < K; k0 += BK) {
        __syncthreads();
        {
            int gr = bM + r;
            uint4 d0 = {0,0,0,0}, d1 = {0,0,0,0};
            if (gr < M) {
                const uint4* src = (const uint4*)(A + (size_t)gr * K + k0 + seg);
                d0 = src[0]; d1 = src[1];
            }
            uint4* dst = (uint4*)(As + r * LDT + seg);
            dst[0] = d0; dst[1] = d1;
        }
        {
            int gr = bN + r;
            uint4 d0 = {0,0,0,0}, d1 = {0,0,0,0};
            if (gr < N) {
                const uint4* src = (const uint4*)(B + (size_t)gr * K + k0 + seg);
                d0 = src[0]; d1 = src[1];
            }
            uint4* dst = (uint4*)(Bs + r * LDT + seg);
            dst[0] = d0; dst[1] = d1;
        }
        __syncthreads();

        Frag16 aF[2], bF[4];
#pragma unroll
        for (int i = 0; i < 2; ++i) {
            int row = waveM * 32 + i * 16 + lr;
            aF[i].q[0] = *(const uint4*)(As + row * LDT + 8 * g);
            aF[i].q[1] = *(const uint4*)(As + row * LDT + 16 + 8 * g);
        }
#pragma unroll
        for (int j = 0; j < 4; ++j) {
            int col = waveN * 64 + j * 16 + lr;
            bF[j].q[0] = *(const uint4*)(Bs + col * LDT + 8 * g);
            bF[j].q[1] = *(const uint4*)(Bs + col * LDT + 16 + 8 * g);
        }
#pragma unroll
        for (int i = 0; i < 2; ++i)
#pragma unroll
            for (int j = 0; j < 4; ++j)
                acc[i][j] = __builtin_amdgcn_wmma_f32_16x16x32_bf16(
                    false, aF[i].v, false, bF[j].v, (short)0, acc[i][j], false, false);
    }
#endif

    // C/D layout: VGPR r -> M = r (lanes 0-15) / M = 8+r (lanes 16-31); N = lane&15
#pragma unroll
    for (int i = 0; i < 2; ++i) {
#pragma unroll
        for (int j = 0; j < 4; ++j) {
            int col = bN + waveN * 64 + j * 16 + lr;
            float bv = (bias != nullptr && col < N) ? bias[col] : 0.0f;
#pragma unroll
            for (int rr = 0; rr < 8; ++rr) {
                int row = bM + waveM * 32 + i * 16 + g * 8 + rr;
                if (row < M && col < N) {
                    float c = acc[i][j][rr] + bv;
                    if (doRelu) c = fmaxf(c, 0.0f);
                    C[(size_t)row * N + col] = c;
                }
            }
        }
    }
}

// --------------------------- elementwise utils -----------------------------
__global__ void f32_to_bf16_k(const float* __restrict__ in,
                              unsigned short* __restrict__ out, size_t n)
{
    size_t i = (size_t)blockIdx.x * blockDim.x + threadIdx.x;
    if (i < n) {
        unsigned int u = __float_as_uint(in[i]);
        out[i] = (unsigned short)((u + 0x7FFFu + ((u >> 16) & 1u)) >> 16); // RNE
    }
}

__global__ void fill_f32_k(float* __restrict__ p, float v, size_t n)
{
    size_t i = (size_t)blockIdx.x * blockDim.x + threadIdx.x;
    if (i < n) p[i] = v;
}

__global__ __launch_bounds__(256)
void bias_relu_k(const float* __restrict__ acc, const float* __restrict__ bias,
                 float* __restrict__ out, size_t total, int H)
{
    size_t i = (size_t)blockIdx.x * blockDim.x + threadIdx.x;
    if (i < total) {
        float v = acc[i] + bias[(int)(i & (size_t)(H - 1))];
        out[i] = fmaxf(v, 0.0f);
    }
}

// row dot with attention vector: al[i] = sum_h X[i,h]*a[h]  (wave per row)
__global__ __launch_bounds__(256)
void rowdot_k(const float* __restrict__ X, const float* __restrict__ a,
              float* __restrict__ out, int nrows, int H)
{
    int row  = blockIdx.x * 8 + (threadIdx.x >> 5);
    int lane = threadIdx.x & 31;
    if (row >= nrows) return;
    float s = 0.0f;
    for (int h = lane; h < H; h += 32) s += X[(size_t)row * H + h] * a[h];
    for (int off = 16; off > 0; off >>= 1) s += __shfl_xor(s, off, 32);
    if (lane == 0) out[row] = s;
}

// ------------------------------ edge passes --------------------------------
__device__ inline void atomicMaxF(float* addr, float val)
{
    if (val >= 0.0f) atomicMax((int*)addr, __float_as_int(val));
    else             atomicMin((unsigned int*)addr, __float_as_uint(val));
}

__global__ void edge_logit_max_k(const int* __restrict__ src, const int* __restrict__ dst,
                                 const float* __restrict__ al_s, const float* __restrict__ al_d,
                                 float* __restrict__ elog, float* __restrict__ m, int E)
{
    int e = blockIdx.x * blockDim.x + threadIdx.x;
    if (e >= E) return;
    float x = al_s[src[e]] + al_d[dst[e]];
    x = (x > 0.0f) ? x : NEG_SLOPE * x;          // leaky_relu
    elog[e] = x;
    atomicMaxF(&m[dst[e]], x);
}

__global__ void edge_expsum_k(const int* __restrict__ dst, float* __restrict__ elog,
                              const float* __restrict__ m, float* __restrict__ s, int E)
{
    int e = blockIdx.x * blockDim.x + threadIdx.x;
    if (e >= E) return;
    int d = dst[e];
    float w = __expf(elog[e] - m[d]);
    elog[e] = w;
    unsafeAtomicAdd(&s[d], w);                   // global_atomic_add_f32
}

// one wave per edge: acc[dst,:] += hs[src,:] * alpha   (8 coalesced sweeps)
__global__ __launch_bounds__(256)
void edge_scatter_k(const int* __restrict__ src, const int* __restrict__ dst,
                    const float* __restrict__ w, const float* __restrict__ ssum,
                    const float* __restrict__ hs, float* __restrict__ acc,
                    int E, int H)
{
    int e = blockIdx.x * 8 + (threadIdx.x >> 5);
    if (e >= E) return;
    int lane = threadIdx.x & 31;
    int si = src[e], di = dst[e];
    float alpha = w[e] / ssum[di];
    const float* hrow = hs  + (size_t)si * H;
    float*       arow = acc + (size_t)di * H;
    for (int h = lane; h < H; h += 32)
        unsafeAtomicAdd(&arow[h], hrow[h] * alpha);
}

// ------------------------------- host side ---------------------------------
static inline dim3 gemm_grid(int M, int N)
{
    return dim3((unsigned)((N + BN - 1) / BN), (unsigned)((M + BM - 1) / BM));
}

struct ConvP {
    const float *Wsrc, *Wdst, *a_src, *a_dst, *bias;
    int Wsrc_n, Wdst_n;
    unsigned short *Wsrc_bf, *Wdst_bf;
};

static void run_conv(hipStream_t stream,
                     const unsigned short* xsrc_bf, int n_src, int k_src,
                     const unsigned short* xdst_bf, int n_dst, int k_dst,
                     const ConvP& p,
                     const int* src, const int* dst, int E,
                     const unsigned short* zpad,
                     float* hs, float* hd, float* al_s, float* al_d,
                     float* m, float* ssum, float* elog, float* acc)
{
    gemm_bf16_wmma<<<gemm_grid(n_src, HIDC), 256, 0, stream>>>(
        xsrc_bf, p.Wsrc_bf, hs, nullptr, zpad, n_src, HIDC, k_src, 0);
    gemm_bf16_wmma<<<gemm_grid(n_dst, HIDC), 256, 0, stream>>>(
        xdst_bf, p.Wdst_bf, hd, nullptr, zpad, n_dst, HIDC, k_dst, 0);
    rowdot_k<<<(n_src + 7) / 8, 256, 0, stream>>>(hs, p.a_src, al_s, n_src, HIDC);
    rowdot_k<<<(n_dst + 7) / 8, 256, 0, stream>>>(hd, p.a_dst, al_d, n_dst, HIDC);
    fill_f32_k<<<(n_dst + 255) / 256, 256, 0, stream>>>(m, -INFINITY, (size_t)n_dst);
    fill_f32_k<<<(n_dst + 255) / 256, 256, 0, stream>>>(ssum, 0.0f, (size_t)n_dst);
    size_t accN = (size_t)n_dst * HIDC;
    fill_f32_k<<<(unsigned)((accN + 255) / 256), 256, 0, stream>>>(acc, 0.0f, accN);
    edge_logit_max_k<<<(E + 255) / 256, 256, 0, stream>>>(src, dst, al_s, al_d, elog, m, E);
    edge_expsum_k<<<(E + 255) / 256, 256, 0, stream>>>(dst, elog, m, ssum, E);
    edge_scatter_k<<<(E + 7) / 8, 256, 0, stream>>>(src, dst, elog, ssum, hs, acc, E, HIDC);
}

extern "C" void kernel_launch(void* const* d_in, const int* in_sizes, int n_in,
                              void* d_out, int out_size, void* d_ws, size_t ws_size,
                              hipStream_t stream)
{
    const float* x_gene = (const float*)d_in[0];
    const float* x_var  = (const float*)d_in[1];
    const int*   src_vg = (const int*)d_in[2];
    const int*   dst_vg = (const int*)d_in[3];
    const int*   src_gv = (const int*)d_in[4];
    const int*   dst_gv = (const int*)d_in[5];

    const int D_GENE = 256, D_VAR = 128, OUTD = 128;
    const int n_gene = in_sizes[0] / D_GENE;   // 20000
    const int n_var  = in_sizes[1] / D_VAR;    // 50000
    const int E      = in_sizes[2];            // 500000

    // ---- params: disambiguate vg/gv ordering via layer-0 W_src shape ----
    const int base = 6;
    const bool vg_first = (in_sizes[base] == HIDC * D_VAR);  // vg W_src: 256x128

    ConvP cp[2][2];   // [layer][0 = vg (var->gene), 1 = gv (gene->var)]
    for (int l = 0; l < 2; ++l) {
        for (int t = 0; t < 2; ++t) {
            int conv = vg_first ? t : 1 - t;
            int b = base + l * 10 + t * 5;
            cp[l][conv].Wsrc   = (const float*)d_in[b + 0];
            cp[l][conv].Wdst   = (const float*)d_in[b + 1];
            cp[l][conv].a_src  = (const float*)d_in[b + 2];
            cp[l][conv].a_dst  = (const float*)d_in[b + 3];
            cp[l][conv].bias   = (const float*)d_in[b + 4];
            cp[l][conv].Wsrc_n = in_sizes[b + 0];
            cp[l][conv].Wdst_n = in_sizes[b + 1];
        }
    }
    const float* WoutG = (const float*)d_in[base + 20];
    const float* bG    = (const float*)d_in[base + 21];
    const float* WoutV = (const float*)d_in[base + 22];
    const float* bV    = (const float*)d_in[base + 23];

    // ---- workspace bump allocator (deterministic) ----
    size_t off = 0;
    auto alloc = [&](size_t bytes) -> void* {
        void* p = (char*)d_ws + off;
        off += (bytes + 255) & ~(size_t)255;
        return p;
    };

    unsigned short* zpad = (unsigned short*)alloc(256);   // zero pad for async OOB reads

    for (int l = 0; l < 2; ++l)
        for (int c = 0; c < 2; ++c) {
            cp[l][c].Wsrc_bf = (unsigned short*)alloc((size_t)cp[l][c].Wsrc_n * 2);
            cp[l][c].Wdst_bf = (unsigned short*)alloc((size_t)cp[l][c].Wdst_n * 2);
        }
    unsigned short* WoutG_bf = (unsigned short*)alloc((size_t)OUTD * HIDC * 2);
    unsigned short* WoutV_bf = (unsigned short*)alloc((size_t)OUTD * HIDC * 2);

    unsigned short* xg_bf = (unsigned short*)alloc((size_t)n_gene * HIDC * 2);
    unsigned short* xv_bf = (unsigned short*)alloc((size_t)n_var  * HIDC * 2);

    float* buf_big   = (float*)alloc((size_t)n_var  * HIDC * 4); // vg hs / gv hd
    float* buf_small = (float*)alloc((size_t)n_gene * HIDC * 4); // vg hd / gv hs
    float* acc_g = (float*)alloc((size_t)n_gene * HIDC * 4);
    float* acc_v = (float*)alloc((size_t)n_var  * HIDC * 4);
    float* xg_f  = (float*)alloc((size_t)n_gene * HIDC * 4);
    float* xv_f  = (float*)alloc((size_t)n_var  * HIDC * 4);
    float* al_src = (float*)alloc((size_t)n_var * 4);
    float* al_dst = (float*)alloc((size_t)n_var * 4);
    float* m_seg  = (float*)alloc((size_t)n_var * 4);
    float* s_seg  = (float*)alloc((size_t)n_var * 4);
    float* elog   = (float*)alloc((size_t)E * 4);
    (void)ws_size; (void)n_in; (void)out_size;

    auto cvt = [&](const float* in, unsigned short* out, size_t n) {
        f32_to_bf16_k<<<(unsigned)((n + 255) / 256), 256, 0, stream>>>(in, out, n);
    };

    // zero the async OOB pad (64 floats == 256B)
    fill_f32_k<<<1, 64, 0, stream>>>((float*)zpad, 0.0f, 64);

    // ---- convert all weights once ----
    for (int l = 0; l < 2; ++l)
        for (int c = 0; c < 2; ++c) {
            cvt(cp[l][c].Wsrc, cp[l][c].Wsrc_bf, (size_t)cp[l][c].Wsrc_n);
            cvt(cp[l][c].Wdst, cp[l][c].Wdst_bf, (size_t)cp[l][c].Wdst_n);
        }
    cvt(WoutG, WoutG_bf, (size_t)OUTD * HIDC);
    cvt(WoutV, WoutV_bf, (size_t)OUTD * HIDC);

    // ---- layers ----
    for (int l = 0; l < 2; ++l) {
        const float* xg_src = (l == 0) ? x_gene : xg_f;
        const float* xv_src = (l == 0) ? x_var  : xv_f;
        const int Kg = D_GENE;                    // 256 both layers
        const int Kv = (l == 0) ? D_VAR : HIDC;   // 128 then 256

        cvt(xg_src, xg_bf, (size_t)n_gene * Kg);
        cvt(xv_src, xv_bf, (size_t)n_var * Kv);

        // vg: var -> gene
        run_conv(stream, xv_bf, n_var, Kv, xg_bf, n_gene, Kg, cp[l][0],
                 src_vg, dst_vg, E, zpad,
                 buf_big, buf_small, al_src, al_dst, m_seg, s_seg, elog, acc_g);
        // gv: gene -> var (reads the same OLD features; buffers reused sequentially)
        run_conv(stream, xg_bf, n_gene, Kg, xv_bf, n_var, Kv, cp[l][1],
                 src_gv, dst_gv, E, zpad,
                 buf_small, buf_big, al_src, al_dst, m_seg, s_seg, elog, acc_v);

        size_t tg = (size_t)n_gene * HIDC, tv = (size_t)n_var * HIDC;
        bias_relu_k<<<(unsigned)((tg + 255) / 256), 256, 0, stream>>>(acc_g, cp[l][0].bias, xg_f, tg, HIDC);
        bias_relu_k<<<(unsigned)((tv + 255) / 256), 256, 0, stream>>>(acc_v, cp[l][1].bias, xv_f, tv, HIDC);
    }

    // ---- output heads ----
    cvt(xg_f, xg_bf, (size_t)n_gene * HIDC);
    cvt(xv_f, xv_bf, (size_t)n_var * HIDC);
    float* out = (float*)d_out;
    gemm_bf16_wmma<<<gemm_grid(n_gene, OUTD), 256, 0, stream>>>(
        xg_bf, WoutG_bf, out, bG, zpad, n_gene, OUTD, HIDC, 0);
    gemm_bf16_wmma<<<gemm_grid(n_var, OUTD), 256, 0, stream>>>(
        xv_bf, WoutV_bf, out + (size_t)n_gene * OUTD, bV, zpad, n_var, OUTD, HIDC, 0);
}
